// Head_438086664803
// MI455X (gfx1250) — matrix operand order repaired and checked by reading, not compiled, not run on hardware
//
#include <hip/hip_runtime.h>
#include <stdint.h>

#define B_BATCH 4
#define T_SEQ   2048
#define C_DIM   1024
#define H_DIM   2048

// LDS padded strides (avoid 64-bank conflicts on row-parallel access)
#define S_STRIDE (T_SEQ + 4)    // floats: 8208 B per row -> 4-bank stagger
#define Q_STRIDE (H_DIM + 8)    // halves: 4112 B per row -> 4-bank stagger

typedef __attribute__((ext_vector_type(16))) __bf16   v16bf;
typedef __attribute__((ext_vector_type(8)))  float    v8f;
typedef __attribute__((ext_vector_type(4)))  float    v4f;
typedef __attribute__((ext_vector_type(4)))  uint32_t v4u;

union Frag {
    v16bf    h;
    v4u      u[2];
    uint16_t s[16];
};

static __device__ __forceinline__ uint16_t f2bf(float f) {
    uint32_t u = __builtin_bit_cast(uint32_t, f);
    u += 0x7FFFu + ((u >> 16) & 1u);          // round-to-nearest-even
    return (uint16_t)(u >> 16);
}

// ---------------------------------------------------------------- convert ---
__global__ void cvt_f32_bf16(const float* __restrict__ src,
                             uint16_t* __restrict__ dst, int n) {
    int i = blockIdx.x * blockDim.x + threadIdx.x;
    int stride = gridDim.x * blockDim.x;
    for (; i < n; i += stride) dst[i] = f2bf(src[i]);
}

// ------------------------------------------------------------- projections ---
// One wave per 16x16 output tile; computes K, Q, V tiles together (A reused).
// K, Q stored row-major [B*T, H] (bf16); V stored transposed [B, H, T] (bf16).
__global__ __launch_bounds__(256)
void proj_kernel(const uint16_t* __restrict__ xb,
                 const uint16_t* __restrict__ wkb,
                 const uint16_t* __restrict__ wqb,
                 const uint16_t* __restrict__ wvb,
                 uint16_t* __restrict__ Kb,
                 uint16_t* __restrict__ Qb,
                 uint16_t* __restrict__ Vtb) {
    const int lane = threadIdx.x & 31;
    const int wave = threadIdx.x >> 5;
    const int tile = blockIdx.x * 8 + wave;      // 512*128 = 65536 tiles
    const int mt = tile >> 7;                    // row tile (B*T/16 = 512)
    const int nt = tile & 127;                   // col tile (H/16 = 128)
    const int r0 = mt * 16;
    const int n0 = nt * 16;
    const int lrow = lane & 15;
    const int lhi  = lane >> 4;

    v8f ck = {0.f,0.f,0.f,0.f,0.f,0.f,0.f,0.f};
    v8f cq = ck, cv = ck;

    const uint16_t* arow = xb  + (size_t)(r0 + lrow) * C_DIM;
    const uint16_t* bkp  = wkb + (size_t)(n0 + lrow) * C_DIM + lhi * 16;
    const uint16_t* bqp  = wqb + (size_t)(n0 + lrow) * C_DIM + lhi * 16;
    const uint16_t* bvp  = wvb + (size_t)(n0 + lrow) * C_DIM + lhi * 16;

    for (int kc = 0; kc < C_DIM / 32; ++kc) {
        Frag a, fk, fq, fv;
        const int k0 = kc * 32 + lhi * 16;
        a.u[0]  = *(const v4u*)(arow + k0);
        a.u[1]  = *(const v4u*)(arow + k0 + 8);
        fk.u[0] = *(const v4u*)(bkp + kc * 32);
        fk.u[1] = *(const v4u*)(bkp + kc * 32 + 8);
        fq.u[0] = *(const v4u*)(bqp + kc * 32);
        fq.u[1] = *(const v4u*)(bqp + kc * 32 + 8);
        fv.u[0] = *(const v4u*)(bvp + kc * 32);
        fv.u[1] = *(const v4u*)(bvp + kc * 32 + 8);
        ck = __builtin_amdgcn_wmma_f32_16x16x32_bf16(false, a.h, false, fk.h, (short)0, ck, false, false);
        cq = __builtin_amdgcn_wmma_f32_16x16x32_bf16(false, a.h, false, fq.h, (short)0, cq, false, false);
        cv = __builtin_amdgcn_wmma_f32_16x16x32_bf16(false, a.h, false, fv.h, (short)0, cv, false, false);
    }

    // K, Q row-major stores (D layout: row = g + 8*lhi, col = lrow)
    for (int g = 0; g < 8; ++g) {
        const int m = g + 8 * lhi;
        const size_t idx = (size_t)(r0 + m) * H_DIM + n0 + lrow;
        Kb[idx] = f2bf(ck[g]);
        Qb[idx] = f2bf(cq[g]);
    }
    // V transposed store: Vt[b][h][t], 8 consecutive t per lane -> one b128
    const int b     = r0 / T_SEQ;
    const int tbase = r0 % T_SEQ;
    v4u pv;
    for (int j = 0; j < 4; ++j)
        pv[j] = (uint32_t)f2bf(cv[2*j]) | ((uint32_t)f2bf(cv[2*j+1]) << 16);
    const size_t vidx = ((size_t)b * H_DIM + n0 + lrow) * T_SEQ + tbase + 8 * lhi;
    *(v4u*)(Vtb + vidx) = pv;
}

// ---------------------------------------------------------------- attention ---
// One workgroup (16 waves) per (batch, 16-row query tile).
__global__ __launch_bounds__(512)
void attn_kernel(const uint16_t* __restrict__ Qb,
                 const uint16_t* __restrict__ Kb,
                 const uint16_t* __restrict__ Vtb,
                 float* __restrict__ out) {
    __shared__ __align__(16) float    S [16 * S_STRIDE];  // ~128 KB scores/probs
    __shared__ __align__(16) uint16_t Qs[16 * Q_STRIDE];  // ~64 KB staged Q tile

    const int tid  = threadIdx.x;
    const int lane = tid & 31;
    const int wave = tid >> 5;                   // 0..15
    const int lrow = lane & 15;
    const int lhi  = lane >> 4;
    const int qt = blockIdx.x & 127;             // query tile within batch
    const int b  = blockIdx.x >> 7;
    const int t0 = qt * 16;
    const int nst = qt + 1;                      // causal: key tiles 0..qt
    const float scale = 0.022097086912079608f;   // 1/sqrt(H)

    // ---- Stage Q tile (16 x H bf16) into LDS: 4096 x b128 copies ----
    for (int i = tid; i < 16 * (H_DIM / 8); i += 512) {
        const int row = i >> 8;                  // 256 v4u per row
        const int c8  = i & 255;
        v4u d = *(const v4u*)(Qb + (size_t)(b * T_SEQ + t0 + row) * H_DIM + c8 * 8);
        *(v4u*)(Qs + row * Q_STRIDE + c8 * 8) = d;
    }
    __syncthreads();

    // ---- Phase 1: scores S[m][s] = scale * Q K^T, causal-masked ----
    const uint16_t* qsr = Qs + lrow * Q_STRIDE;  // A row base in LDS
    for (int st = wave; st < nst; st += 16) {
        v8f acc = {0.f,0.f,0.f,0.f,0.f,0.f,0.f,0.f};
        const uint16_t* krow = Kb + (size_t)(b * T_SEQ + st * 16 + lrow) * H_DIM + lhi * 16;
        for (int kc = 0; kc < H_DIM / 32; ++kc) {
            Frag a, bb;
            const int k0 = kc * 32 + lhi * 16;
            a.u[0]  = *(const v4u*)(qsr + k0);
            a.u[1]  = *(const v4u*)(qsr + k0 + 8);
            bb.u[0] = *(const v4u*)(krow + kc * 32);
            bb.u[1] = *(const v4u*)(krow + kc * 32 + 8);
            acc = __builtin_amdgcn_wmma_f32_16x16x32_bf16(false, a.h, false, bb.h, (short)0, acc, false, false);
        }
        for (int g = 0; g < 8; ++g) {
            const int m = g + 8 * lhi;
            const int s = st * 16 + lrow;
            float v = acc[g] * scale;
            v = (s <= t0 + m) ? v : -INFINITY;
            S[m * S_STRIDE + s] = v;
        }
    }
    __syncthreads();

    // ---- Phase 2: per-row softmax (wave w owns row w), zero the pad ----
    {
        const int row = wave;
        const int nS = nst * 16;
        const int nSpad = ((nS + 31) >> 5) << 5;
        float* Sr = S + row * S_STRIDE;
        float mx = -INFINITY;
        for (int s = lane; s < nS; s += 32) mx = fmaxf(mx, Sr[s]);
        for (int off = 16; off > 0; off >>= 1) mx = fmaxf(mx, __shfl_xor(mx, off, 32));
        float sum = 0.f;
        for (int s = lane; s < nS; s += 32) {
            float e = __expf(Sr[s] - mx);
            sum += e;
            Sr[s] = e;
        }
        for (int off = 16; off > 0; off >>= 1) sum += __shfl_xor(sum, off, 32);
        const float inv = 1.0f / sum;
        for (int s = lane; s < nSpad; s += 32)
            Sr[s] = (s < nS) ? Sr[s] * inv : 0.f;
    }
    __syncthreads();

    // ---- Phase 3: O = P V ; wave w owns columns [w*128, w*128+128) ----
    {
        const int nS = nst * 16;
        const int nch = (nS + 31) >> 5;          // 32-wide K chunks
        const int h0 = wave * 128;
        v8f acc[8];
        for (int i = 0; i < 8; ++i) acc[i] = (v8f){0.f,0.f,0.f,0.f,0.f,0.f,0.f,0.f};

        for (int sc = 0; sc < nch; ++sc) {
            Frag a;
            const float* pr = S + lrow * S_STRIDE + sc * 32 + lhi * 16;
            v4f p0 = *(const v4f*)(pr);
            v4f p1 = *(const v4f*)(pr + 4);
            v4f p2 = *(const v4f*)(pr + 8);
            v4f p3 = *(const v4f*)(pr + 12);
            for (int j = 0; j < 4; ++j) {
                a.s[j]      = f2bf(p0[j]);
                a.s[4 + j]  = f2bf(p1[j]);
                a.s[8 + j]  = f2bf(p2[j]);
                a.s[12 + j] = f2bf(p3[j]);
            }
            for (int ntl = 0; ntl < 8; ++ntl) {
                Frag bv;
                const uint16_t* vr = Vtb + ((size_t)b * H_DIM + h0 + ntl * 16 + lrow) * T_SEQ
                                   + sc * 32 + lhi * 16;
                bv.u[0] = *(const v4u*)(vr);
                bv.u[1] = *(const v4u*)(vr + 8);
                acc[ntl] = __builtin_amdgcn_wmma_f32_16x16x32_bf16(false, a.h, false, bv.h, (short)0, acc[ntl], false, false);
            }
        }
        for (int ntl = 0; ntl < 8; ++ntl)
            for (int g = 0; g < 8; ++g) {
                const int m = g + 8 * lhi;
                out[((size_t)(b * T_SEQ) + t0 + m) * H_DIM + h0 + ntl * 16 + lrow] = acc[ntl][g];
            }
    }
}

// ---------------------------------------------------------------- launcher ---
extern "C" void kernel_launch(void* const* d_in, const int* in_sizes, int n_in,
                              void* d_out, int out_size, void* d_ws, size_t ws_size,
                              hipStream_t stream) {
    const float* x  = (const float*)d_in[0];
    const float* Wk = (const float*)d_in[1];
    const float* Wq = (const float*)d_in[2];
    const float* Wv = (const float*)d_in[3];
    float* out = (float*)d_out;

    const size_t NX = (size_t)B_BATCH * T_SEQ * C_DIM;   // 8,388,608
    const size_t NW = (size_t)H_DIM * C_DIM;             // 2,097,152
    const size_t NP = (size_t)B_BATCH * T_SEQ * H_DIM;   // 16,777,216

    char* ws = (char*)d_ws;
    uint16_t* xb  = (uint16_t*)(ws);
    uint16_t* wkb = (uint16_t*)(ws + NX * 2);
    uint16_t* wqb = (uint16_t*)(ws + NX * 2 + NW * 2);
    uint16_t* wvb = (uint16_t*)(ws + NX * 2 + NW * 4);
    uint16_t* Qb  = (uint16_t*)(ws + NX * 2 + NW * 6);
    uint16_t* Kb  = (uint16_t*)(ws + NX * 2 + NW * 6 + NP * 2);
    uint16_t* Vtb = (uint16_t*)(ws + NX * 2 + NW * 6 + NP * 4);

    cvt_f32_bf16<<<4096, 256, 0, stream>>>(x,  xb,  (int)NX);
    cvt_f32_bf16<<<2048, 256, 0, stream>>>(Wk, wkb, (int)NW);
    cvt_f32_bf16<<<2048, 256, 0, stream>>>(Wq, wqb, (int)NW);
    cvt_f32_bf16<<<2048, 256, 0, stream>>>(Wv, wvb, (int)NW);

    // 65536 tiles, 8 waves per block
    proj_kernel<<<8192, 256, 0, stream>>>(xb, wkb, wqb, wvb, Kb, Qb, Vtb);

    // one workgroup per (batch, query tile)
    attn_kernel<<<B_BATCH * (T_SEQ / 16), 512, 0, stream>>>(Qb, Kb, Vtb, out);
}
